// AAF_GraphSAGE_Conv_32804960207312
// MI455X (gfx1250) — matrix operand order repaired and checked by reading, not compiled, not run on hardware
//
#include <hip/hip_runtime.h>
#include <hip/hip_bf16.h>

typedef __attribute__((ext_vector_type(16))) __bf16 v16bf;
typedef __attribute__((ext_vector_type(8)))  float  v8f;
typedef __attribute__((ext_vector_type(4)))  float  f32x4;

constexpr int Nn   = 100000;
constexpr int Ee   = 1600000;
constexpr int FIN  = 128;
constexpr int Hh   = 256;
constexpr int Cc   = 40;
constexpr float BN_EPS = 1e-5f;

// ---------------------------------------------------------------------------
// f32 -> bf16 weight conversion
// ---------------------------------------------------------------------------
__global__ void __launch_bounds__(256)
f32_to_bf16_kernel(const float* __restrict__ src, __bf16* __restrict__ dst, int n) {
    int i = blockIdx.x * 256 + threadIdx.x;
    if (i < n) dst[i] = (__bf16)src[i];
}

__global__ void __launch_bounds__(256)
fill_zero_kernel(float* __restrict__ p, int n) {
    int i = blockIdx.x * 256 + threadIdx.x;
    if (i < n) p[i] = 0.0f;
}

// ---------------------------------------------------------------------------
// Edge scatter: one wave per edge, lane handles 8 contiguous features.
// agg[dst] += h[src]; cnt[dst] += 1
// ---------------------------------------------------------------------------
__global__ void __launch_bounds__(256)
scatter_add_kernel(const float* __restrict__ h, const long long* __restrict__ ei,
                   float* __restrict__ agg, float* __restrict__ cnt) {
    int e    = blockIdx.x * 8 + (threadIdx.x >> 5);
    int lane = threadIdx.x & 31;
    long long src = ei[e];
    long long dst = ei[(long long)Ee + e];
    const float* hs = h   + (size_t)src * Hh + lane * 8;
    float*       ad = agg + (size_t)dst * Hh + lane * 8;
    f32x4 v0 = *(const f32x4*)(hs);
    f32x4 v1 = *(const f32x4*)(hs + 4);
#pragma unroll
    for (int j = 0; j < 4; ++j)
        __hip_atomic_fetch_add(ad + j, v0[j], __ATOMIC_RELAXED, __HIP_MEMORY_SCOPE_AGENT);
#pragma unroll
    for (int j = 0; j < 4; ++j)
        __hip_atomic_fetch_add(ad + 4 + j, v1[j], __ATOMIC_RELAXED, __HIP_MEMORY_SCOPE_AGENT);
    if (lane == 0)
        __hip_atomic_fetch_add(cnt + dst, 1.0f, __ATOMIC_RELAXED, __HIP_MEMORY_SCOPE_AGENT);
}

// agg <- agg / max(cnt,1)   (in place, 25.6M elements)
__global__ void __launch_bounds__(256)
mean_kernel(float* __restrict__ agg, const float* __restrict__ cnt) {
    int i = blockIdx.x * 256 + threadIdx.x;
    float c = cnt[i >> 8];
    agg[i] = agg[i] / fmaxf(c, 1.0f);
}

// ---------------------------------------------------------------------------
// WMMA bf16 GEMM (+ optional second GEMM) + bias + eval-BatchNorm + ReLU.
// out[m, 0:256] = relu(BN( A1[m,:K] @ B1^T + (A2 @ B2^T) + bias ))
// B is W[n,k] row-major (i.e. lin.weight), so per-lane reads are contiguous.
// One wave computes a 16x16 tile; 16 n-tiles per m-tile.
// ---------------------------------------------------------------------------
__device__ __forceinline__ v16bf load_a_frag(const float* __restrict__ A,
                                             int row, int K, int k0, int lane) {
    // A 16x32 layout (matches B): lanes 0-15 -> K = k0+[0..15]
    //                             lanes16-31 -> K = k0+16+[0..15]
    const float* p = A + (size_t)row * K + k0 + ((lane & 16) ? 16 : 0);
    v16bf r;
#pragma unroll
    for (int e = 0; e < 8; ++e) r[e]     = (__bf16)p[e];
#pragma unroll
    for (int e = 0; e < 8; ++e) r[8 + e] = (__bf16)p[8 + e];
    return r;
}

__device__ __forceinline__ v16bf load_b_frag(const __bf16* __restrict__ W,
                                             int col, int K, int k0, int lane) {
    // B 32x16 layout, contiguous k (matches A): lane = column, lanes 16-31 hold K+16..K+31.
    const __bf16* p = W + (size_t)col * K + k0 + ((lane & 16) ? 16 : 0);
    return *(const v16bf*)p;  // 32B contiguous, aligned
}

__global__ void __launch_bounds__(256)
wmma_gemm_bn_relu_kernel(const float* __restrict__ A1, const __bf16* __restrict__ B1,
                         const float* __restrict__ A2, const __bf16* __restrict__ B2,
                         const float* __restrict__ bias, const float* __restrict__ bn,
                         float* __restrict__ out, int K) {
    int gwave  = (blockIdx.x * 256 + threadIdx.x) >> 5;
    int lane   = threadIdx.x & 31;
    int tile_n = gwave & 15;          // 256/16 = 16 n-tiles
    int tile_m = gwave >> 4;
    int rowA   = tile_m * 16 + (lane & 15);
    int col    = tile_n * 16 + (lane & 15);

    v8f acc = {0.f, 0.f, 0.f, 0.f, 0.f, 0.f, 0.f, 0.f};
    for (int k0 = 0; k0 < K; k0 += 32) {
        v16bf a = load_a_frag(A1, rowA, K, k0, lane);
        v16bf b = load_b_frag(B1, col, K, k0, lane);
        acc = __builtin_amdgcn_wmma_f32_16x16x32_bf16(false, a, false, b,
                                                      (short)0, acc, false, false);
    }
    if (A2 != nullptr) {              // uniform branch: EXEC stays all-ones
        for (int k0 = 0; k0 < Hh; k0 += 32) {
            v16bf a = load_a_frag(A2, rowA, Hh, k0, lane);
            v16bf b = load_b_frag(B2, col, Hh, k0, lane);
            acc = __builtin_amdgcn_wmma_f32_16x16x32_bf16(false, a, false, b,
                                                          (short)0, acc, false, false);
        }
    }

    // eval BatchNorm folded: y = (x - rm)*g*rsqrt(rv+eps) + beta
    float g  = bn[col];
    float bt = bn[256 + col];
    float rm = bn[512 + col];
    float rv = bn[768 + col];
    float scale = g * rsqrtf(rv + BN_EPS);
    float shift = bt - rm * scale;
    float bsv   = bias[col];

    int mbase = tile_m * 16 + ((lane & 16) ? 8 : 0);
#pragma unroll
    for (int r = 0; r < 8; ++r) {
        float t = (acc[r] + bsv) * scale + shift;
        out[(size_t)(mbase + r) * Hh + col] = fmaxf(t, 0.0f);
    }
}

// ---------------------------------------------------------------------------
// Head: logits = h @ post2_w^T + b  ->  log_softmax.  Weights in LDS (40 KB).
// ---------------------------------------------------------------------------
__global__ void __launch_bounds__(256)
post2_logsoftmax_kernel(const float* __restrict__ h, const float* __restrict__ w,
                        const float* __restrict__ b, float* __restrict__ out) {
    __shared__ float wsh[Cc * Hh];
    __shared__ float bsh[Cc];
    for (int t = threadIdx.x; t < Cc * Hh; t += 256) wsh[t] = w[t];
    if (threadIdx.x < Cc) bsh[threadIdx.x] = b[threadIdx.x];
    __syncthreads();

    int i = blockIdx.x * 256 + threadIdx.x;
    if (i >= Nn) return;

    float acc[Cc];
#pragma unroll
    for (int c = 0; c < Cc; ++c) acc[c] = bsh[c];

    const float* hr = h + (size_t)i * Hh;
    for (int k = 0; k < Hh; ++k) {
        float hk = hr[k];
#pragma unroll
        for (int c = 0; c < Cc; ++c) acc[c] = fmaf(hk, wsh[c * Hh + k], acc[c]);
    }
    float mx = acc[0];
#pragma unroll
    for (int c = 1; c < Cc; ++c) mx = fmaxf(mx, acc[c]);
    float s = 0.0f;
#pragma unroll
    for (int c = 0; c < Cc; ++c) s += __expf(acc[c] - mx);
    float ls = __logf(s) + mx;
#pragma unroll
    for (int c = 0; c < Cc; ++c) out[(size_t)i * Cc + c] = acc[c] - ls;
}

// ---------------------------------------------------------------------------
extern "C" void kernel_launch(void* const* d_in, const int* in_sizes, int n_in,
                              void* d_out, int out_size, void* d_ws, size_t ws_size,
                              hipStream_t stream) {
    const float*     x        = (const float*)d_in[0];
    const long long* ei       = (const long long*)d_in[1];
    const float*     pre_w    = (const float*)d_in[2];
    const float*     pre_b    = (const float*)d_in[3];
    const float*     bnp      = (const float*)d_in[4];   // [5,4,256]
    const float*     lin_l_w  = (const float*)d_in[5];   // [3,256,256]
    const float*     lin_l_b  = (const float*)d_in[6];   // [3,256]
    const float*     lin_r_w  = (const float*)d_in[7];   // [3,256,256]
    const float*     post1_w  = (const float*)d_in[8];
    const float*     post1_b  = (const float*)d_in[9];
    const float*     post2_w  = (const float*)d_in[10];
    const float*     post2_b  = (const float*)d_in[11];

    char*  ws  = (char*)d_ws;
    size_t off = 0;
    auto alloc = [&](size_t bytes) -> void* {
        void* p = ws + off;
        off = (off + bytes + 255) & ~(size_t)255;
        return p;
    };
    float*  hA    = (float*)alloc((size_t)Nn * Hh * 4);
    float*  hB    = (float*)alloc((size_t)Nn * Hh * 4);
    float*  agg   = (float*)alloc((size_t)Nn * Hh * 4);
    float*  cnt   = (float*)alloc((size_t)Nn * 4);
    __bf16* preWb = (__bf16*)alloc((size_t)Hh * FIN * 2);
    __bf16* linlB = (__bf16*)alloc((size_t)3 * Hh * Hh * 2);
    __bf16* linrB = (__bf16*)alloc((size_t)3 * Hh * Hh * 2);
    __bf16* p1Wb  = (__bf16*)alloc((size_t)Hh * Hh * 2);

    auto cvt = [&](const float* s, __bf16* d, int n) {
        f32_to_bf16_kernel<<<(n + 255) / 256, 256, 0, stream>>>(s, d, n);
    };
    cvt(pre_w,   preWb, Hh * FIN);
    cvt(lin_l_w, linlB, 3 * Hh * Hh);
    cvt(lin_r_w, linrB, 3 * Hh * Hh);
    cvt(post1_w, p1Wb,  Hh * Hh);

    const int gemm_blocks = (Nn / 16) * (Hh / 16) / 8;   // 12500 blocks, 8 waves each

    // pre layer: h = relu(BN0(x @ pre_w^T + pre_b))
    wmma_gemm_bn_relu_kernel<<<gemm_blocks, 256, 0, stream>>>(
        x, preWb, nullptr, nullptr, pre_b, bnp + 0 * 4 * Hh, hA, FIN);

    float* cur = hA;
    float* nxt = hB;
    for (int i = 0; i < 3; ++i) {
        fill_zero_kernel<<<(Nn * Hh) / 256, 256, 0, stream>>>(agg, Nn * Hh);
        fill_zero_kernel<<<(Nn + 255) / 256, 256, 0, stream>>>(cnt, Nn);
        scatter_add_kernel<<<Ee / 8, 256, 0, stream>>>(cur, ei, agg, cnt);
        mean_kernel<<<(Nn * Hh) / 256, 256, 0, stream>>>(agg, cnt);
        wmma_gemm_bn_relu_kernel<<<gemm_blocks, 256, 0, stream>>>(
            agg, linlB + (size_t)i * Hh * Hh,
            cur, linrB + (size_t)i * Hh * Hh,
            lin_l_b + (size_t)i * Hh, bnp + (size_t)(i + 1) * 4 * Hh, nxt, Hh);
        float* t = cur; cur = nxt; nxt = t;
    }

    // post1: h = relu(BN4(h @ post1_w^T + post1_b))
    wmma_gemm_bn_relu_kernel<<<gemm_blocks, 256, 0, stream>>>(
        cur, p1Wb, nullptr, nullptr, post1_b, bnp + 4 * 4 * Hh, nxt, Hh);

    // head + log_softmax -> d_out  [N, 40] f32
    post2_logsoftmax_kernel<<<(Nn + 255) / 256, 256, 0, stream>>>(
        nxt, post2_w, post2_b, (float*)d_out);
}
